// MultiHeadAttention_24919400252136
// MI455X (gfx1250) — compile-verified, matrix operand order checked by reading, not run on hardware
//
#include <hip/hip_runtime.h>
#include <hip/hip_bf16.h>

// ---------------------------------------------------------------------------
// MI455X (gfx1250) multi-head attention forward, bf16 WMMA + flash softmax.
// B=8, N=1024, E=768, H=12, D=64.
// - All LDS operand tiles are laid out so every WMMA fragment read is two
//   contiguous 16-byte runs per lane -> ds_load_b128 pairs (no scalar gathers).
// - Flash kernel computes S^T = K.Q^T so that (a) softmax rows are per-lane
//   (1 shuffle per reduction) and (b) the S^T C-fragments ARE the P A-fragments
//   after an in-register f32->bf16 cast (no LDS round trip for P).
// ---------------------------------------------------------------------------

typedef __attribute__((ext_vector_type(16))) __bf16 v16bf;
typedef __attribute__((ext_vector_type(8)))  float  v8f;
typedef __attribute__((ext_vector_type(4)))  __bf16 v4bf;

#define WMMA_BF16(a, b, c) \
  __builtin_amdgcn_wmma_f32_16x16x32_bf16(false, (a), false, (b), (short)0, (c), false, false)

// K-index mapping for 16-bit A/B fragments (ISA 7.12.2, 16-bit 16x32):
// per lane-half h: i=0..7 -> K = i+8h ; i=8..15 -> K = 16+(i-8)+8h
// (two contiguous 8-element runs -> merges to 2x ds_load_b128)
__device__ __forceinline__ int kmapw(int i, int h) {
  int j = i >> 1, p = i & 1;
  int base = (j < 4) ? (2 * j + p) : (16 + 2 * (j - 4) + p);
  return base + 8 * h;
}

// A fragment (16 rows x 32 K) from row-major LDS tile [rows][stride]
__device__ __forceinline__ v16bf ldfrag_a(const __bf16* lds, int row0, int k0, int stride) {
  int lane = threadIdx.x & 31, m = lane & 15, h = lane >> 4;
  v16bf a;
#pragma unroll
  for (int i = 0; i < 16; ++i) a[i] = lds[(row0 + m) * stride + k0 + kmapw(i, h)];
  return a;
}

// B fragment (32 K x 16 cols) from COLUMN-major LDS tile [cols][stride]
// element (kk, n) = lds[(col0+n)*stride + k0 + kk]   -> contiguous K per lane
__device__ __forceinline__ v16bf ldfrag_b(const __bf16* lds, int k0, int col0, int stride) {
  int lane = threadIdx.x & 31, n = lane & 15, h = lane >> 4;
  v16bf b;
#pragma unroll
  for (int i = 0; i < 16; ++i) b[i] = lds[(col0 + n) * stride + k0 + kmapw(i, h)];
  return b;
}

// ---------------------------------------------------------------------------
// Kernel 1: qkv = x @ w_qkv + b_qkv, scattered to Q/K/V [B,H,N,D] bf16.
// Block tile 128(M) x 128(N), BK=32, 256 threads = 8 waves.
// Wave grid 4(M-strips of 32) x 2(N-strips of 64); 2x4 WMMA tiles per wave.
// ---------------------------------------------------------------------------
__global__ __launch_bounds__(256) void mha_qkv_gemm(
    const float* __restrict__ x, const float* __restrict__ w,
    const float* __restrict__ bias,
    __bf16* __restrict__ Qb, __bf16* __restrict__ Kb, __bf16* __restrict__ Vb) {
  constexpr int KDIM = 768, NW = 2304, KT = KDIM / 32;
  __shared__ __bf16 As[128][40];    // [row][k]
  __shared__ __bf16 BsT[128][40];   // [col][k]  (transposed at staging)

  const int tid = threadIdx.x;
  const int wave = tid >> 5, wr = wave & 3, wc = wave >> 2;
  const int rowBase = blockIdx.y * 128;
  const int colBase = blockIdx.x * 128;

  v8f acc[2][4];
#pragma unroll
  for (int i = 0; i < 2; ++i)
#pragma unroll
    for (int j = 0; j < 4; ++j) acc[i][j] = (v8f){};

  const float4* xg = reinterpret_cast<const float4*>(x);
  const float4* wg = reinterpret_cast<const float4*>(w);

  for (int kt = 0; kt < KT; ++kt) {
    __syncthreads();
#pragma unroll
    for (int l = 0; l < 4; ++l) {  // A: 128x32 f32 -> bf16, vector b64 stores
      int idx = tid + l * 256, r = idx >> 3, c4 = idx & 7;
      float4 v = xg[((size_t)(rowBase + r) * KDIM + kt * 32 + c4 * 4) >> 2];
      v4bf t = {(__bf16)v.x, (__bf16)v.y, (__bf16)v.z, (__bf16)v.w};
      *reinterpret_cast<v4bf*>(&As[r][c4 * 4]) = t;
    }
#pragma unroll
    for (int l = 0; l < 4; ++l) {  // B: 32x128 f32 -> bf16, transposed stores
      int idx = tid + l * 256, r = idx >> 5, c4 = idx & 31;  // r = k, c4 = col/4
      float4 v = wg[((size_t)(kt * 32 + r) * NW + colBase + c4 * 4) >> 2];
      BsT[c4 * 4 + 0][r] = (__bf16)v.x;
      BsT[c4 * 4 + 1][r] = (__bf16)v.y;
      BsT[c4 * 4 + 2][r] = (__bf16)v.z;
      BsT[c4 * 4 + 3][r] = (__bf16)v.w;
    }
    if (kt + 1 < KT) {  // CDNA5 global_prefetch_b8 for next k-tile
      __builtin_prefetch(x + (size_t)(rowBase + (tid >> 1)) * KDIM + (kt + 1) * 32 + (tid & 1) * 16, 0, 1);
      __builtin_prefetch(w + (size_t)(kt * 32 + 32 + (tid >> 3)) * NW + colBase + (tid & 7) * 16, 0, 1);
    }
    __syncthreads();

    v16bf a0 = ldfrag_a(&As[0][0], wr * 32, 0, 40);
    v16bf a1 = ldfrag_a(&As[0][0], wr * 32 + 16, 0, 40);
#pragma unroll
    for (int tj = 0; tj < 4; ++tj) {
      v16bf b = ldfrag_b(&BsT[0][0], 0, wc * 64 + tj * 16, 40);
      acc[0][tj] = WMMA_BF16(a0, b, acc[0][tj]);
      acc[1][tj] = WMMA_BF16(a1, b, acc[1][tj]);
    }
  }

  // epilogue: bias add + scatter. Column c = h*(D*3) + d*3 + s  (qkv innermost).
  const int lane = tid & 31, n = lane & 15, h = lane >> 4;
  __bf16* bufs[3] = {Qb, Kb, Vb};
#pragma unroll
  for (int ti = 0; ti < 2; ++ti)
#pragma unroll
    for (int tj = 0; tj < 4; ++tj)
#pragma unroll
      for (int r = 0; r < 8; ++r) {
        int m = rowBase + wr * 32 + ti * 16 + r + 8 * h;
        int c = colBase + wc * 64 + tj * 16 + n;
        float val = acc[ti][tj][r] + bias[c];
        int s = c % 3, cd = c / 3, d = cd & 63, hh = cd >> 6;
        int bidx = m >> 10, nidx = m & 1023;
        bufs[s][(((size_t)(bidx * 12 + hh) << 10) + nidx) * 64 + d] = (__bf16)val;
      }
}

// ---------------------------------------------------------------------------
// Kernel 2: flash attention per (b,h,64-row q tile). 128 threads = 4 waves,
// wave w owns q-rows [16w,16w+16). Computes S^T = K.Q^T so each lane owns one
// q-row for the softmax stats; S^T C-frags are recast in-register into the
// P A-fragments for O += P.V. Post-softmax /sqrt(E). Output attn [B,N,E] bf16.
// ---------------------------------------------------------------------------
__global__ __launch_bounds__(128) void mha_flash_attn(
    const __bf16* __restrict__ Qb, const __bf16* __restrict__ Kb,
    const __bf16* __restrict__ Vb, __bf16* __restrict__ attn) {
  __shared__ __bf16 Qs[64][72];    // [q-row][d]  (col-major for S^T's B operand)
  __shared__ __bf16 Ks[64][72];    // [key][d]    (row-major for S^T's A operand)
  __shared__ __bf16 VsT[64][72];   // [d][key]    (col-major for PV's B operand)

  const int tid = threadIdx.x;
  const int wave = tid >> 5, lane = tid & 31, n = lane & 15, h = lane >> 4;
  const int bh = blockIdx.y, b = bh / 12, head = bh % 12;
  const int q0 = blockIdx.x * 64;
  const size_t base = (size_t)bh * 1024 * 64;

  // load Q tile (64x64 bf16)
  const v4bf* qg = reinterpret_cast<const v4bf*>(Qb + base + (size_t)q0 * 64);
#pragma unroll
  for (int l = 0; l < 8; ++l) {
    int idx = tid + l * 128, r = idx >> 4, c4 = idx & 15;
    *reinterpret_cast<v4bf*>(&Qs[r][c4 * 4]) = qg[r * 16 + c4];
  }
  __syncthreads();
  // B operand of S^T: (d, qrow) -> Qs[qrow][d]; loop-invariant
  v16bf qb0 = ldfrag_b(&Qs[0][0], 0, wave * 16, 72);
  v16bf qb1 = ldfrag_b(&Qs[0][0], 32, wave * 16, 72);

  // per-lane softmax state: this lane's q-row is (wave*16 + n)
  float m_run = -1e30f, l_run = 0.0f;
  v8f o[4];
#pragma unroll
  for (int dt = 0; dt < 4; ++dt) o[dt] = (v8f){};

  for (int kt = 0; kt < 16; ++kt) {
    __syncthreads();  // protect prior-iteration reads of Ks/VsT
    const v4bf* kg = reinterpret_cast<const v4bf*>(Kb + base + (size_t)kt * 64 * 64);
    const v4bf* vg = reinterpret_cast<const v4bf*>(Vb + base + (size_t)kt * 64 * 64);
#pragma unroll
    for (int l = 0; l < 8; ++l) {
      int idx = tid + l * 128, r = idx >> 4, c4 = idx & 15;  // r = key, c4 = d/4
      *reinterpret_cast<v4bf*>(&Ks[r][c4 * 4]) = kg[r * 16 + c4];
      v4bf vv = vg[r * 16 + c4];
      VsT[c4 * 4 + 0][r] = vv[0];
      VsT[c4 * 4 + 1][r] = vv[1];
      VsT[c4 * 4 + 2][r] = vv[2];
      VsT[c4 * 4 + 3][r] = vv[3];
    }
    __syncthreads();

    // S^T tile (64 keys x 16 qrows): sT[ct] VGPR r = key ct*16 + r + 8h,
    // lane n = q-row wave*16+n. Contraction over d=64 in 2 WMMA steps.
    v8f sT[4];
#pragma unroll
    for (int ct = 0; ct < 4; ++ct) {
      v8f a = (v8f){};
      a = WMMA_BF16(ldfrag_a(&Ks[0][0], ct * 16, 0, 72), qb0, a);
      a = WMMA_BF16(ldfrag_a(&Ks[0][0], ct * 16, 32, 72), qb1, a);
      sT[ct] = a;
    }

    // online softmax: this lane holds 32 of its q-row's 64 scores (key-half h)
    float mx = sT[0][0];
#pragma unroll
    for (int ct = 0; ct < 4; ++ct)
#pragma unroll
      for (int r = 0; r < 8; ++r) mx = fmaxf(mx, sT[ct][r]);
    mx = fmaxf(mx, __shfl_xor(mx, 16, 32));  // combine key halves
    float newm = fmaxf(m_run, mx);
    float alpha = __expf(m_run - newm);
    float sum = 0.0f;
#pragma unroll
    for (int ct = 0; ct < 4; ++ct)
#pragma unroll
      for (int r = 0; r < 8; ++r) {
        sT[ct][r] = __expf(sT[ct][r] - newm);
        sum += sT[ct][r];
      }
    sum += __shfl_xor(sum, 16, 32);
    l_run = l_run * alpha + sum;
    m_run = newm;

    // rescale O: row of o-frag element r is q-row r+8h -> broadcast alpha
#pragma unroll
    for (int r = 0; r < 8; ++r) {
      float av = __shfl(alpha, r + 8 * h, 32);
#pragma unroll
      for (int dt = 0; dt < 4; ++dt) o[dt][r] *= av;
    }

    // P A-fragments directly from S^T C-frags (pure in-register casts):
    // pa element i<8  -> K=i+8h    == sT[tile0][i]
    // pa element i>=8 -> K=16+(i-8)+8h == sT[tile1][i-8]
    v16bf pa0, pa1;
#pragma unroll
    for (int i = 0; i < 8; ++i) {
      pa0[i]     = (__bf16)sT[0][i];
      pa0[i + 8] = (__bf16)sT[1][i];
      pa1[i]     = (__bf16)sT[2][i];
      pa1[i + 8] = (__bf16)sT[3][i];
    }

    // O += P . V  (contraction over 64 keys in 2 steps)
#pragma unroll
    for (int dt = 0; dt < 4; ++dt) {
      o[dt] = WMMA_BF16(pa0, ldfrag_b(&VsT[0][0], 0, dt * 16, 72), o[dt]);
      o[dt] = WMMA_BF16(pa1, ldfrag_b(&VsT[0][0], 32, dt * 16, 72), o[dt]);
    }
  }

  // epilogue: divide by rowsum * sqrt(E) (post-softmax scaling per reference)
  const float scaling = 27.712812921102035f;  // sqrt(768)
#pragma unroll
  for (int r = 0; r < 8; ++r) {
    float lv = __shfl(l_run, r + 8 * h, 32);
    float inv = 1.0f / (lv * scaling);
#pragma unroll
    for (int dt = 0; dt < 4; ++dt) {
      int mg = q0 + wave * 16 + r + 8 * h;
      float val = o[dt][r] * inv;
      attn[(size_t)(b * 1024 + mg) * 768 + head * 64 + dt * 16 + n] = (__bf16)val;
    }
  }
}

// ---------------------------------------------------------------------------
// Kernel 3: out = attn @ w_proj + b_proj  (f32 output). Same tiling as K1.
// ---------------------------------------------------------------------------
__global__ __launch_bounds__(256) void mha_proj_gemm(
    const __bf16* __restrict__ A, const float* __restrict__ w,
    const float* __restrict__ bias, float* __restrict__ out) {
  constexpr int KDIM = 768, NW = 768, KT = KDIM / 32;
  __shared__ __bf16 As[128][40];    // [row][k]
  __shared__ __bf16 BsT[128][40];   // [col][k]

  const int tid = threadIdx.x;
  const int wave = tid >> 5, wr = wave & 3, wc = wave >> 2;
  const int rowBase = blockIdx.y * 128;
  const int colBase = blockIdx.x * 128;

  v8f acc[2][4];
#pragma unroll
  for (int i = 0; i < 2; ++i)
#pragma unroll
    for (int j = 0; j < 4; ++j) acc[i][j] = (v8f){};

  const v4bf* ag = reinterpret_cast<const v4bf*>(A);
  const float4* wg = reinterpret_cast<const float4*>(w);

  for (int kt = 0; kt < KT; ++kt) {
    __syncthreads();
#pragma unroll
    for (int l = 0; l < 4; ++l) {  // A: 128x32 bf16, vector b64 copies
      int idx = tid + l * 256, r = idx >> 3, c4 = idx & 7;
      *reinterpret_cast<v4bf*>(&As[r][c4 * 4]) =
          ag[((size_t)(rowBase + r) * KDIM + kt * 32 + c4 * 4) >> 2];
    }
#pragma unroll
    for (int l = 0; l < 4; ++l) {  // B: 32x128 f32 -> bf16, transposed stores
      int idx = tid + l * 256, r = idx >> 5, c4 = idx & 31;
      float4 v = wg[((size_t)(kt * 32 + r) * NW + colBase + c4 * 4) >> 2];
      BsT[c4 * 4 + 0][r] = (__bf16)v.x;
      BsT[c4 * 4 + 1][r] = (__bf16)v.y;
      BsT[c4 * 4 + 2][r] = (__bf16)v.z;
      BsT[c4 * 4 + 3][r] = (__bf16)v.w;
    }
    if (kt + 1 < KT) {
      __builtin_prefetch(w + (size_t)(kt * 32 + 32 + (tid >> 3)) * NW + colBase + (tid & 7) * 16, 0, 1);
    }
    __syncthreads();

    v16bf a0 = ldfrag_a(&As[0][0], wr * 32, 0, 40);
    v16bf a1 = ldfrag_a(&As[0][0], wr * 32 + 16, 0, 40);
#pragma unroll
    for (int tj = 0; tj < 4; ++tj) {
      v16bf b = ldfrag_b(&BsT[0][0], 0, wc * 64 + tj * 16, 40);
      acc[0][tj] = WMMA_BF16(a0, b, acc[0][tj]);
      acc[1][tj] = WMMA_BF16(a1, b, acc[1][tj]);
    }
  }

  const int lane = tid & 31, n = lane & 15, h = lane >> 4;
#pragma unroll
  for (int ti = 0; ti < 2; ++ti)
#pragma unroll
    for (int tj = 0; tj < 4; ++tj)
#pragma unroll
      for (int r = 0; r < 8; ++r) {
        int m = rowBase + wr * 32 + ti * 16 + r + 8 * h;
        int c = colBase + wc * 64 + tj * 16 + n;
        out[(size_t)m * NW + c] = acc[ti][tj][r] + bias[c];
      }
}

// ---------------------------------------------------------------------------
extern "C" void kernel_launch(void* const* d_in, const int* in_sizes, int n_in,
                              void* d_out, int out_size, void* d_ws, size_t ws_size,
                              hipStream_t stream) {
  (void)in_sizes; (void)n_in; (void)out_size; (void)ws_size;
  const float* x      = (const float*)d_in[0];
  const float* w_qkv  = (const float*)d_in[1];
  const float* b_qkv  = (const float*)d_in[2];
  const float* w_proj = (const float*)d_in[3];
  const float* b_proj = (const float*)d_in[4];
  float* out = (float*)d_out;

  // ws layout: Q | K | V | attn, each 8*12*1024*64 = 6,291,456 bf16 (~50 MB total)
  const size_t qkvElems = (size_t)8 * 12 * 1024 * 64;
  __bf16* Qb   = (__bf16*)d_ws;
  __bf16* Kb   = Qb + qkvElems;
  __bf16* Vb   = Kb + qkvElems;
  __bf16* attn = Vb + qkvElems;

  // 1) qkv projection: [8192 x 768] x [768 x 2304]
  mha_qkv_gemm<<<dim3(2304 / 128, 8192 / 128), 256, 0, stream>>>(x, w_qkv, b_qkv, Qb, Kb, Vb);
  // 2) flash attention: grid (q-tiles, B*H)
  mha_flash_attn<<<dim3(1024 / 64, 8 * 12), 128, 0, stream>>>(Qb, Kb, Vb, attn);
  // 3) output projection: [8192 x 768] x [768 x 768]
  mha_proj_gemm<<<dim3(768 / 128, 8192 / 128), 256, 0, stream>>>(attn, w_proj, b_proj, out);
}
